// DoubleALIFRNN_89756226552397
// MI455X (gfx1250) — compile-verified, matrix operand order checked by reading, not run on hardware
//
#include <hip/hip_runtime.h>
#include <cmath>

// ---------------- problem constants ----------------
constexpr int kSEQ   = 1000;
constexpr int kBATCH = 256;
constexpr int kIN    = 700;
constexpr int kH1    = 256;
constexpr int kH2    = 256;
constexpr int kOUT   = 20;
constexpr float kB0   = 0.01f;
constexpr float kBETA = 1.8f;

// ---------------- mapping constants ----------------
constexpr int kNWG   = 16;          // persistent workgroups; 128 waves x (16x32 tile) = 256 tiles/layer
constexpr int kKX    = 704;         // x K padded 700 -> 704 (22 * 32)
constexpr int kKPAD1 = kKX + kH1;   // 960: [x | z1] K extent for layer-1 weight slice

// dynamic LDS: W1 slice 32x960 + W2 slice 32x512 + Wout slice 32x256, fp16
constexpr int kW1E   = 32 * kKPAD1;        // 30720 elems
constexpr int kW2E   = 32 * 512;           // 16384 elems
constexpr int kWOE   = 32 * 256;           //  8192 elems
constexpr int kSMEM  = (kW1E + kW2E + kWOE) * 2;   // 110592 bytes (of 320 KB/WGP)

typedef __attribute__((ext_vector_type(16))) _Float16 v16h;
typedef __attribute__((ext_vector_type(8)))  float    v8f;

__device__ __forceinline__ v8f zero8() {
  v8f z;
#pragma unroll
  for (int i = 0; i < 8; ++i) z[i] = 0.0f;
  return z;
}

__device__ __forceinline__ v8f wmma_f16(v16h a, v16h b, v8f c) {
  // D = A(16x32 f16) * B(32x16 f16) + C(16x16 f32)
  return __builtin_amdgcn_wmma_f32_16x16x32_f16(false, a, false, b, (short)0, c,
                                                false, false);
}

// ---- A-fragment (16x32 f16) from a row-major fp32 matrix row, per ISA layout:
// lane L: M = L&15; lanes 0-15 hold K = k0+[0..7] and k0+16+[0..7],
// lanes 16-31 hold K = k0+8+[0..7] and k0+24+[0..7].
__device__ __forceinline__ v16h afrag_f32(const float* __restrict__ row, int k0, int lane) {
  const int kb = (lane & 16) ? 8 : 0;
  v16h a;
#pragma unroll
  for (int j = 0; j < 8; ++j) {
    a[j]     = (_Float16)row[k0 + kb + j];
    a[8 + j] = (_Float16)row[k0 + 16 + kb + j];
  }
  return a;
}

__device__ __forceinline__ v16h afrag_f32_guard(const float* __restrict__ row, int k0,
                                                int kmax, int lane) {
  const int kb = (lane & 16) ? 8 : 0;
  v16h a;
#pragma unroll
  for (int j = 0; j < 8; ++j) {
    const int k1 = k0 + kb + j, k2 = k0 + 16 + kb + j;
    a[j]     = (k1 < kmax) ? (_Float16)row[k1] : (_Float16)0.0f;
    a[8 + j] = (k2 < kmax) ? (_Float16)row[k2] : (_Float16)0.0f;
  }
  return a;
}

// A- or B-fragment from contiguous fp16 (LDS weight slice row, or global spike row).
__device__ __forceinline__ v16h frag_f16(const _Float16* __restrict__ row, int k0, int lane) {
  const int kb = (lane & 16) ? 8 : 0;
  v16h a;
#pragma unroll
  for (int j = 0; j < 8; ++j) {
    a[j]     = row[k0 + kb + j];
    a[8 + j] = row[k0 + 16 + kb + j];
  }
  return a;
}

// Grid barrier: generation counter, acq/rel at agent scope; spinner sleeps.
__device__ __forceinline__ void grid_barrier(unsigned* cnt, unsigned* gen,
                                             unsigned nwg, unsigned& target) {
  __syncthreads();
  if (threadIdx.x == 0) {
    ++target;
    const unsigned prev =
        __hip_atomic_fetch_add(cnt, 1u, __ATOMIC_ACQ_REL, __HIP_MEMORY_SCOPE_AGENT);
    if (prev == nwg - 1u) {
      __hip_atomic_store(cnt, 0u, __ATOMIC_RELAXED, __HIP_MEMORY_SCOPE_AGENT);
      __hip_atomic_fetch_add(gen, 1u, __ATOMIC_RELEASE, __HIP_MEMORY_SCOPE_AGENT);
    } else {
      while (__hip_atomic_load(gen, __ATOMIC_ACQUIRE, __HIP_MEMORY_SCOPE_AGENT) < target)
        __builtin_amdgcn_s_sleep(1);
    }
  }
  __syncthreads();
}

// ---------------- init: exp tables, zero barrier + spike ping-pong buffers --------------
__global__ void alif_init(const float* __restrict__ tm1, const float* __restrict__ ta1,
                          const float* __restrict__ tm2, const float* __restrict__ ta2,
                          const float* __restrict__ tout,
                          float* alpha1, float* rho1, float* alpha2, float* rho2,
                          float* aout, unsigned* bar,
                          _Float16* z1buf, _Float16* z2buf) {
  const int gid = blockIdx.x * blockDim.x + threadIdx.x;
  const int nth = gridDim.x * blockDim.x;
  if (gid < kH1) {
    alpha1[gid] = expf(-1.0f / tm1[gid]);
    rho1[gid]   = expf(-1.0f / ta1[gid]);
  } else if (gid < 2 * kH1) {
    const int i = gid - kH1;
    alpha2[i] = expf(-1.0f / tm2[i]);
    rho2[i]   = expf(-1.0f / ta2[i]);
  } else if (gid < 2 * kH1 + 32) {
    const int i = gid - 2 * kH1;
    aout[i] = (i < kOUT) ? expf(-1.0f / tout[i]) : 0.0f;
  }
  if (gid < 2) bar[gid] = 0u;
  for (int i = gid; i < 2 * kBATCH * kH1; i += nth) z1buf[i] = (_Float16)0.0f;
  for (int i = gid; i < 2 * kBATCH * kH2; i += nth) z2buf[i] = (_Float16)0.0f;
}

// ---------------- persistent recurrent kernel ----------------
__global__ __launch_bounds__(256, 1)
void alif_persistent(const float* __restrict__ x, const float* __restrict__ W1,
                     const float* __restrict__ W2, const float* __restrict__ Wout,
                     const float* __restrict__ alpha1, const float* __restrict__ rho1,
                     const float* __restrict__ alpha2, const float* __restrict__ rho2,
                     const float* __restrict__ aout,
                     _Float16* __restrict__ z1buf,   // 2 * [kBATCH][kH1] fp16 ping-pong
                     _Float16* __restrict__ z2buf,   // 2 * [kBATCH][kH2]
                     unsigned* __restrict__ bar,
                     float* __restrict__ out) {
  extern __shared__ _Float16 smem[];
  _Float16* W1s = smem;                 // [32][kKPAD1]  this WG's 32 H1 rows, K=[x(704)|z1(256)]
  _Float16* W2s = smem + kW1E;          // [32][512]     32 H2 rows, K=[z1 | z2]
  _Float16* Wos = smem + kW1E + kW2E;   // [32][256]     32 out rows (zero-padded past 20)

  const int tid  = threadIdx.x;
  const int lane = tid & 31;
  const int wave = tid >> 5;
  const int wg   = blockIdx.x;

  const int npair  = wg & 7;                 // 32-column slice of N (layers 1 & 2)
  const int npbase = npair * 32;
  const int mtile  = (wg & 8) + wave;        // batch M tile, 0..15

  // ---- stage weights into LDS as fp16 ----
  for (int idx = tid; idx < kW1E; idx += 256) {
    const int n = idx / kKPAD1, k = idx % kKPAD1;
    const int nrow = npbase + n;
    float w = 0.0f;
    if (k < kIN)       w = W1[nrow * (kIN + kH1) + k];
    else if (k >= kKX) w = W1[nrow * (kIN + kH1) + kIN + (k - kKX)];
    W1s[idx] = (_Float16)w;
  }
  for (int idx = tid; idx < kW2E; idx += 256) {
    const int n = idx >> 9, k = idx & 511;
    W2s[idx] = (_Float16)W2[(npbase + n) * (kH1 + kH2) + k];
  }
  for (int idx = tid; idx < kWOE; idx += 256) {
    const int n = idx >> 8, k = idx & 255;
    Wos[idx] = (_Float16)((n < kOUT) ? Wout[n * kH2 + k] : 0.0f);
  }
  __syncthreads();

  const int nl    = lane & 15;
  const int rbase = (lane & 16) ? 8 : 0;     // C/D layout: lanes 16-31 hold M = r+8
  const int rowA  = mtile * 16 + nl;         // batch row this lane serves in A-fragments
  const int ncol0 = npbase + nl;             // neuron column (sub-tile 0) in C/D
  const int ncol1 = npbase + 16 + nl;        // neuron column (sub-tile 1)
  const int ocol  = wave * 16 + nl;          // readout column (waves 0,1 only)

  const float al1c[2] = {alpha1[ncol0], alpha1[ncol1]};
  const float rh1c[2] = {rho1[ncol0],   rho1[ncol1]};
  const float al2c[2] = {alpha2[ncol0], alpha2[ncol1]};
  const float rh2c[2] = {rho2[ncol0],   rho2[ncol1]};
  const float om_al1[2] = {1.0f - al1c[0], 1.0f - al1c[1]};
  const float om_rh1[2] = {1.0f - rh1c[0], 1.0f - rh1c[1]};
  const float om_al2[2] = {1.0f - al2c[0], 1.0f - al2c[1]};
  const float om_rh2[2] = {1.0f - rh2c[0], 1.0f - rh2c[1]};
  const float alo    = (wave < 2) ? aout[ocol] : 0.0f;
  const float om_alo = 1.0f - alo;

  const _Float16* __restrict__ w1r0 = &W1s[nl * kKPAD1];
  const _Float16* __restrict__ w1r1 = &W1s[(16 + nl) * kKPAD1];
  const _Float16* __restrict__ w2r0 = &W2s[nl * 512];
  const _Float16* __restrict__ w2r1 = &W2s[(16 + nl) * 512];
  const _Float16* __restrict__ worow = &Wos[(wave * 16 + nl) * 256];

  // ALIF state, register-resident in accumulator layout for the whole sequence.
  v8f u1[2] = {zero8(), zero8()}, a1[2] = {zero8(), zero8()}, z1[2] = {zero8(), zero8()};
  v8f u2[2] = {zero8(), zero8()}, a2[2] = {zero8(), zero8()}, z2[2] = {zero8(), zero8()};
  v8f uo = zero8();

  unsigned target = 0;

  for (int t = 0; t < kSEQ; ++t) {
    const int wr = t & 1, rd = wr ^ 1;

    // ------- layer 1: cur1 = [x_t, z1_prev] @ W1^T, 16x32 tile, paired K-steps -------
    const float* __restrict__ xrow = x + (size_t)t * kBATCH * kIN + (size_t)rowA * kIN;
    __builtin_prefetch(xrow + (size_t)kBATCH * kIN, 0, 1);  // next timestep's row
    v8f acc0 = zero8(), acc1 = zero8();
#pragma unroll 2
    for (int kp = 0; kp < 10; ++kp) {        // K-steps 0..19, two A-frags in flight
      const int k0 = kp * 64, k1 = k0 + 32;
      const v16h aA = afrag_f32(xrow, k0, lane);
      const v16h aB = afrag_f32(xrow, k1, lane);
      acc0 = wmma_f16(aA, frag_f16(w1r0, k0, lane), acc0);
      acc1 = wmma_f16(aA, frag_f16(w1r1, k0, lane), acc1);
      acc0 = wmma_f16(aB, frag_f16(w1r0, k1, lane), acc0);
      acc1 = wmma_f16(aB, frag_f16(w1r1, k1, lane), acc1);
    }
    {                                        // tail: K-step 20 (full) + 21 (guarded)
      const v16h aA = afrag_f32(xrow, 640, lane);
      const v16h aB = afrag_f32_guard(xrow, 672, kIN, lane);
      acc0 = wmma_f16(aA, frag_f16(w1r0, 640, lane), acc0);
      acc1 = wmma_f16(aA, frag_f16(w1r1, 640, lane), acc1);
      acc0 = wmma_f16(aB, frag_f16(w1r0, 672, lane), acc0);
      acc1 = wmma_f16(aB, frag_f16(w1r1, 672, lane), acc1);
    }
    {                                        // recurrent part: z1_prev, 4 pairs
      const _Float16* __restrict__ zrow = z1buf + rd * kBATCH * kH1 + rowA * kH1;
#pragma unroll
      for (int kp = 0; kp < 4; ++kp) {
        const int k0 = kp * 64, k1 = k0 + 32;
        const v16h aA = frag_f16(zrow, k0, lane);
        const v16h aB = frag_f16(zrow, k1, lane);
        acc0 = wmma_f16(aA, frag_f16(w1r0, kKX + k0, lane), acc0);
        acc1 = wmma_f16(aA, frag_f16(w1r1, kKX + k0, lane), acc1);
        acc0 = wmma_f16(aB, frag_f16(w1r0, kKX + k1, lane), acc0);
        acc1 = wmma_f16(aB, frag_f16(w1r1, kKX + k1, lane), acc1);
      }
    }
    {   // ALIF update, layer 1 (spikes exact in fp16)
      _Float16* __restrict__ zw = z1buf + wr * kBATCH * kH1;
#pragma unroll
      for (int h = 0; h < 2; ++h) {
        const v8f acc = h ? acc1 : acc0;
        const int nc = h ? ncol1 : ncol0;
#pragma unroll
        for (int r = 0; r < 8; ++r) {
          a1[h][r] = rh1c[h] * a1[h][r] + om_rh1[h] * z1[h][r];
          const float th = kB0 + kBETA * a1[h][r];
          u1[h][r] = al1c[h] * u1[h][r] + om_al1[h] * acc[r] - z1[h][r] * th;
          z1[h][r] = (u1[h][r] - th) > 0.0f ? 1.0f : 0.0f;
          zw[(mtile * 16 + rbase + r) * kH1 + nc] = (_Float16)z1[h][r];
        }
      }
    }
    grid_barrier(&bar[0], &bar[1], kNWG, target);   // z1_t visible device-wide

    // ------- layer 2: cur2 = [z1_t, z2_prev] @ W2^T, 16x32 tile, paired K-steps -------
    {
      v8f b0 = zero8(), b1 = zero8();
      const _Float16* __restrict__ z1row = z1buf + wr * kBATCH * kH1 + rowA * kH1;
#pragma unroll
      for (int kp = 0; kp < 4; ++kp) {
        const int k0 = kp * 64, k1 = k0 + 32;
        const v16h aA = frag_f16(z1row, k0, lane);
        const v16h aB = frag_f16(z1row, k1, lane);
        b0 = wmma_f16(aA, frag_f16(w2r0, k0, lane), b0);
        b1 = wmma_f16(aA, frag_f16(w2r1, k0, lane), b1);
        b0 = wmma_f16(aB, frag_f16(w2r0, k1, lane), b0);
        b1 = wmma_f16(aB, frag_f16(w2r1, k1, lane), b1);
      }
      const _Float16* __restrict__ z2row = z2buf + rd * kBATCH * kH2 + rowA * kH2;
#pragma unroll
      for (int kp = 0; kp < 4; ++kp) {
        const int k0 = kp * 64, k1 = k0 + 32;
        const v16h aA = frag_f16(z2row, k0, lane);
        const v16h aB = frag_f16(z2row, k1, lane);
        b0 = wmma_f16(aA, frag_f16(w2r0, 256 + k0, lane), b0);
        b1 = wmma_f16(aA, frag_f16(w2r1, 256 + k0, lane), b1);
        b0 = wmma_f16(aB, frag_f16(w2r0, 256 + k1, lane), b0);
        b1 = wmma_f16(aB, frag_f16(w2r1, 256 + k1, lane), b1);
      }
      _Float16* __restrict__ zw = z2buf + wr * kBATCH * kH2;
#pragma unroll
      for (int h = 0; h < 2; ++h) {
        const v8f acc = h ? b1 : b0;
        const int nc = h ? ncol1 : ncol0;
#pragma unroll
        for (int r = 0; r < 8; ++r) {
          a2[h][r] = rh2c[h] * a2[h][r] + om_rh2[h] * z2[h][r];
          const float th = kB0 + kBETA * a2[h][r];
          u2[h][r] = al2c[h] * u2[h][r] + om_al2[h] * acc[r] - z2[h][r] * th;
          z2[h][r] = (u2[h][r] - th) > 0.0f ? 1.0f : 0.0f;
          zw[(mtile * 16 + rbase + r) * kH2 + nc] = (_Float16)z2[h][r];
        }
      }
    }
    grid_barrier(&bar[0], &bar[1], kNWG, target);   // z2_t visible device-wide

    // -------- readout: uo = ao*uo + (1-ao)*(z2_t @ Wout^T); waves 0,1 own out tiles ------
    if (wave < 2) {
      v8f accE = zero8(), accO = zero8();          // independent even/odd K accumulators
      const _Float16* __restrict__ z2row =
          z2buf + wr * kBATCH * kH2 + (size_t)(wg * 16 + nl) * kH2;
#pragma unroll
      for (int kp = 0; kp < 4; ++kp) {
        const int k0 = kp * 64, k1 = k0 + 32;
        accE = wmma_f16(frag_f16(z2row, k0, lane), frag_f16(worow, k0, lane), accE);
        accO = wmma_f16(frag_f16(z2row, k1, lane), frag_f16(worow, k1, lane), accO);
      }
#pragma unroll
      for (int r = 0; r < 8; ++r) {
        uo[r] = alo * uo[r] + om_alo * (accE[r] + accO[r]);
        if (t >= 1 && ocol < kOUT)
          out[((size_t)(t - 1) * kBATCH + (wg * 16 + rbase + r)) * kOUT + ocol] = uo[r];
      }
    }
  }
}

// ---------------- host entry ----------------
extern "C" void kernel_launch(void* const* d_in, const int* in_sizes, int n_in,
                              void* d_out, int out_size, void* d_ws, size_t ws_size,
                              hipStream_t stream) {
  const float* x    = (const float*)d_in[0];
  const float* W1   = (const float*)d_in[1];
  const float* tm1  = (const float*)d_in[2];
  const float* ta1  = (const float*)d_in[3];
  const float* W2   = (const float*)d_in[4];
  const float* tm2  = (const float*)d_in[5];
  const float* ta2  = (const float*)d_in[6];
  const float* Wout = (const float*)d_in[7];
  const float* tout = (const float*)d_in[8];
  float* out = (float*)d_out;

  // workspace layout (~524 KB spikes + 8 KB tables)
  char* w = (char*)d_ws;
  float*    alpha1 = (float*)(w + 0);
  float*    rho1   = (float*)(w + 1024);
  float*    alpha2 = (float*)(w + 2048);
  float*    rho2   = (float*)(w + 3072);
  float*    aout   = (float*)(w + 4096);
  unsigned* bar    = (unsigned*)(w + 4352);
  _Float16* z1buf  = (_Float16*)(w + 8192);
  _Float16* z2buf  = (_Float16*)(w + 8192 + (size_t)2 * kBATCH * kH1 * sizeof(_Float16));

  hipFuncSetAttribute((const void*)alif_persistent,
                      hipFuncAttributeMaxDynamicSharedMemorySize, kSMEM);

  alif_init<<<dim3(512), dim3(256), 0, stream>>>(tm1, ta1, tm2, ta2, tout, alpha1, rho1,
                                                 alpha2, rho2, aout, bar, z1buf, z2buf);
  alif_persistent<<<dim3(kNWG), dim3(256), kSMEM, stream>>>(
      x, W1, W2, Wout, alpha1, rho1, alpha2, rho2, aout, z1buf, z2buf, bar, out);
}
